// TypeEnsembleSheafLearner_31842887533252
// MI455X (gfx1250) — matrix operand order repaired and checked by reading, not compiled, not run on hardware
//
#include <hip/hip_runtime.h>
#include <math.h>

// ---------------------------------------------------------------------------
// TypeEnsembleSheafLearner for MI455X (gfx1250, wave32, WMMA)
//
// out[e] = tanh( concat(x[src[e]], x[dst[e]]) @ W[type[e]] ),  W[t]: [256,16]
//
// Strategy: bucket edges by type -> grouped GEMM, one wave per 16-edge tile,
// K=256 reduced with 8x v_wmma_f32_16x16x32_f16 (A gathered from f16 copy of
// x, B pre-packed per type into the WMMA fragment layout).
// ---------------------------------------------------------------------------

typedef _Float16 v16h __attribute__((ext_vector_type(16)));
typedef _Float16 v8h  __attribute__((ext_vector_type(8)));
typedef float    v8f  __attribute__((ext_vector_type(8)));

#define C_DIM       128   // node feature dim
#define TWO_C       256   // 2*C
#define P_DIM       16    // 4x4 restriction map
#define T_TYPES     12
#define SLICES      8     // 256 / 32 (K per WMMA)

// meta[] int layout in workspace
#define MT_CNT    0   // 12 counts
#define MT_CUR    12  // 12 scatter cursors
#define MT_EDGE   24  // 13 edge-offset prefix
#define MT_TILE   37  // 13 tile-offset prefix
#define MT_TOTAL  50  // total tiles
#define MT_WORDS  64

// ---------------------------------------------------------------- init meta
__global__ void k_init_meta(int* meta) {
    if (threadIdx.x < MT_WORDS) meta[threadIdx.x] = 0;
}

// ------------------------------------------------------------- x f32 -> f16
__global__ void k_convert_x(const float* __restrict__ x,
                            _Float16* __restrict__ xh, int n) {
    int i = blockIdx.x * blockDim.x + threadIdx.x;
    if (i < n) xh[i] = (_Float16)x[i];
}

// ---------------------------------------------- pack W into B-fragment form
// B fragment for slice s of type t, lane L, element i (16 halves per lane):
//   K = s*32 + (L/16)*16 + i ,  N = L%16
// stored at bfrag[ ((t*8 + s)*32 + L)*16 + i ]
__global__ void k_pack_w(const float* __restrict__ W,
                         _Float16* __restrict__ bfrag) {
    int t    = blockIdx.x;            // 12 blocks
    int tid  = threadIdx.x;           // 256 threads -> (s, lane)
    int s    = tid >> 5;
    int lane = tid & 31;
    int n    = lane & 15;
    int kb   = s * 32 + (lane >> 4) * 16;
    const float* Wt = W + (size_t)t * TWO_C * P_DIM;
    _Float16* dst = bfrag + ((size_t)((t * SLICES + s) * 32 + lane)) * 16;
#pragma unroll
    for (int i = 0; i < 16; ++i)
        dst[i] = (_Float16)Wt[(size_t)(kb + i) * P_DIM + n];
}

// ------------------------------------------------------------ count by type
__global__ void k_count(const int* __restrict__ types, int* meta, int E) {
    int e = blockIdx.x * blockDim.x + threadIdx.x;
    if (e < E) atomicAdd(&meta[MT_CNT + types[e]], 1);
}

// -------------------------------------------------------- tiny prefix scans
__global__ void k_scan(int* meta) {
    int eacc = 0, tacc = 0;
#pragma unroll
    for (int t = 0; t < T_TYPES; ++t) {
        meta[MT_EDGE + t] = eacc;
        meta[MT_TILE + t] = tacc;
        int c = meta[MT_CNT + t];
        eacc += c;
        tacc += (c + 15) >> 4;
    }
    meta[MT_EDGE + T_TYPES] = eacc;
    meta[MT_TILE + T_TYPES] = tacc;
    meta[MT_TOTAL] = tacc;
}

// --------------------------------------------------- scatter edges to bucket
__global__ void k_scatter(const int* __restrict__ types, int* meta,
                          int* __restrict__ bucket, int E) {
    int e = blockIdx.x * blockDim.x + threadIdx.x;
    if (e < E) {
        int t = types[e];
        int pos = meta[MT_EDGE + t] + atomicAdd(&meta[MT_CUR + t], 1);
        bucket[pos] = e;
    }
}

// -------------------------------------------------------------- grouped GEMM
// 8 waves per block; each wave owns one 16-edge tile of a single type.
__global__ __launch_bounds__(256) void k_gemm(
    const _Float16* __restrict__ xh,
    const _Float16* __restrict__ bfrag,
    const int* __restrict__ edge_index,   // [2, E] (src row then dst row)
    const int* __restrict__ meta,
    const int* __restrict__ bucket,
    float* __restrict__ out, int E) {

    const int lane = threadIdx.x & 31;
    const int wave = threadIdx.x >> 5;
    const int tile = blockIdx.x * 8 + wave;
    if (tile >= meta[MT_TOTAL]) return;           // wave-uniform: EXEC stays full

    // locate bucket (type) for this tile
    int t = 0;
    while (t < T_TYPES - 1 && tile >= meta[MT_TILE + t + 1]) ++t;
    const int segStart = meta[MT_EDGE + t];
    const int segEnd   = meta[MT_EDGE + t + 1];
    const int m0       = (tile - meta[MT_TILE + t]) << 4;

    const int hi  = lane >> 4;    // half-wave selects K sub-range
    const int row = lane & 15;    // A-matrix row this lane feeds

    // edge for the row this lane loads (clamp padding lanes to a valid edge
    // so every lane issues real loads -> no divergence before WMMA)
    int epos = segStart + m0 + row;
    int eid  = bucket[(epos < segEnd) ? epos : (segStart + m0)];
    const _Float16* srcRow = xh + (size_t)edge_index[eid] * C_DIM;
    const _Float16* dstRow = xh + (size_t)edge_index[E + eid] * C_DIM;

    // per-lane B pointer for this type
    const _Float16* bf = bfrag + ((size_t)(t * SLICES) * 32 + lane) * 16;

    v8f c = {};
#pragma unroll
    for (int s = 0; s < SLICES; ++s) {
        // slices 0..3 read the src half of x_cat, 4..7 the dst half
        const _Float16* rowp = ((s < 4) ? srcRow : dstRow) + (s & 3) * 32;
        // A operand k order used here (16x32): lane holds K = k0+16*hi..+15 (matches B packing)
        v8h lo = *(const v8h*)(rowp + 16 * hi);
        v8h up = *(const v8h*)(rowp + 16 * hi + 8);
        v16h a;
#pragma unroll
        for (int i = 0; i < 8; ++i) { a[i] = lo[i]; a[8 + i] = up[i]; }
        v16h b = *(const v16h*)(bf + (size_t)s * 32 * 16);
        c = __builtin_amdgcn_wmma_f32_16x16x32_f16(
                /*neg_a=*/false, a, /*neg_b=*/false, b,
                /*c_mod=*/(short)0, c, /*reuse_a=*/false, /*reuse_b=*/false);
    }

    // C/D layout: VGPR j -> M = j + 8*hi, N = lane%16
    const int col = lane & 15;
#pragma unroll
    for (int j = 0; j < 8; ++j) {
        int M  = j + 8 * hi;
        int ep = segStart + m0 + M;
        if (ep < segEnd) {
            int e = bucket[ep];
            out[(size_t)e * P_DIM + col] = tanhf(c[j]);
        }
    }
}

// ---------------------------------------------------------------------------
static inline size_t align256(size_t v) { return (v + 255) & ~(size_t)255; }

extern "C" void kernel_launch(void* const* d_in, const int* in_sizes, int n_in,
                              void* d_out, int out_size, void* d_ws, size_t ws_size,
                              hipStream_t stream) {
    const float* x          = (const float*)d_in[0];   // [N, 128]
    const int*   edge_index = (const int*)d_in[1];     // [2, E]
    const int*   edge_types = (const int*)d_in[2];     // [E]
    const float* W          = (const float*)d_in[3];   // [12, 256, 16]

    const int NC = in_sizes[0];
    const int E  = in_sizes[2];

    // workspace carve-up
    char*  ws = (char*)d_ws;
    size_t o  = 0;
    _Float16* xh    = (_Float16*)(ws + o); o += align256((size_t)NC * sizeof(_Float16));
    _Float16* bfrag = (_Float16*)(ws + o); o += align256((size_t)T_TYPES * SLICES * 32 * 16 * sizeof(_Float16));
    int*      meta  = (int*)(ws + o);      o += align256(MT_WORDS * sizeof(int));
    int*      bucket= (int*)(ws + o);      o += align256((size_t)E * sizeof(int));
    (void)ws_size; (void)n_in; (void)out_size;

    k_init_meta<<<1, 64, 0, stream>>>(meta);
    k_convert_x<<<(NC + 255) / 256, 256, 0, stream>>>(x, xh, NC);
    k_pack_w<<<T_TYPES, 256, 0, stream>>>(W, bfrag);
    k_count<<<(E + 255) / 256, 256, 0, stream>>>(edge_types, meta, E);
    k_scan<<<1, 1, 0, stream>>>(meta);
    k_scatter<<<(E + 255) / 256, 256, 0, stream>>>(edge_types, meta, bucket, E);

    const int maxTiles = E / 16 + T_TYPES;          // upper bound incl. partial tiles
    k_gemm<<<(maxTiles + 7) / 8, 256, 0, stream>>>(xh, bfrag, edge_index, meta,
                                                   bucket, (float*)d_out, E);
}
